// MaskedAttention_3083786519245
// MI455X (gfx1250) — matrix operand order repaired and checked by reading, not compiled, not run on hardware
//
#include <hip/hip_runtime.h>
#include <hip/hip_bf16.h>

// ---------------------------------------------------------------------------
// MaskedAttention for MI455X (gfx1250).
// All three GEMMs via v_wmma_f32_16x16x32_f16, 32x64 C-tile per wave
// (8 accumulators: A fragments reused 4x across N, B fragments reused 2x
// across M). B fragments batch-loaded per K-step so loads stay in flight.
// LDS-tiled transpose for V; fp32 softmax writing the w output in place.
// ---------------------------------------------------------------------------

typedef _Float16 half8_t  __attribute__((ext_vector_type(8)));
typedef _Float16 v16h     __attribute__((ext_vector_type(16)));
typedef float    v8f      __attribute__((ext_vector_type(8)));
typedef float    float8_t __attribute__((ext_vector_type(8)));

#define BDIM 4
#define SDIM 2048
#define DDIM 1024
#define E3D  3072            // 3*D
#define BS   (BDIM * SDIM)   // 8192
#define INV_SCALE 0.03125f   // 1/sqrt(1024)

// ---- WMMA fragment helpers (layouts per CDNA5 ISA 7.12.2) ------------------
// A (16x32 f16): lane&15 = M row; halves 0-7 -> K = kb..kb+7, halves 8-15 ->
// K = kb+16..kb+23, kb = 8*(lane>=16). Row contiguous along K in memory.
__device__ __forceinline__ v16h load_frag_a_f16(const _Float16* __restrict__ row,
                                                int k0, int lane) {
  const int kb = k0 + ((lane >> 4) << 3);
  half8_t lo = *(const half8_t*)(row + kb);
  half8_t hi = *(const half8_t*)(row + kb + 16);
  v16h r;
#pragma unroll
  for (int i = 0; i < 8; ++i) { r[i] = lo[i]; r[i + 8] = hi[i]; }
  return r;
}

// Same A layout, fp32 source converted to f16 in-register.
__device__ __forceinline__ v16h load_frag_a_f32(const float* __restrict__ row,
                                                int k0, int lane) {
  const int kb = k0 + ((lane >> 4) << 3);
  float8_t lo = *(const float8_t*)(row + kb);
  float8_t hi = *(const float8_t*)(row + kb + 16);
  v16h r;
#pragma unroll
  for (int i = 0; i < 8; ++i) { r[i] = (_Float16)lo[i]; r[i + 8] = (_Float16)hi[i]; }
  return r;
}

// B (32x16 f16): lane&15 = N column; halves 0-7 -> K = kb..kb+7, halves 8-15 ->
// K = kb+16..kb+23, kb = k0 + 8*(lane>=16).
__device__ __forceinline__ v16h load_frag_b_f16(const _Float16* __restrict__ colrow,
                                                int k0, int lane) {
  const int kb = k0 + ((lane >> 4) << 3);
  half8_t lo = *(const half8_t*)(colrow + kb);
  half8_t hi = *(const half8_t*)(colrow + kb + 16);
  v16h r;
#pragma unroll
  for (int i = 0; i < 8; ++i) { r[i] = lo[i]; r[i + 8] = hi[i]; }
  return r;
}

__device__ __forceinline__ v8f wmma_f16(v16h a, v16h b, v8f c) {
  return __builtin_amdgcn_wmma_f32_16x16x32_f16(false, a, false, b, (short)0, c,
                                                false, false);
}

// ---------------------------------------------------------------------------
// Kernel 1: fp32 -> f16 conversion (x and W_qkv)
// ---------------------------------------------------------------------------
__global__ void __launch_bounds__(256)
cvt_f32_to_f16_kernel(const float* __restrict__ src, _Float16* __restrict__ dst, int n) {
  int i = blockIdx.x * 256 + threadIdx.x;
  if (i < n) dst[i] = (_Float16)src[i];
}

// ---------------------------------------------------------------------------
// Kernel 2: QKV projection  qkv[m, n] = sum_k x[m,k] * W[n,k] + b[n]
//   M = 8192, N = 3072, K = 1024.  Wave: 32x64 tile.  Block: 8 waves = 128x128.
// ---------------------------------------------------------------------------
__global__ void __launch_bounds__(256)
qkv_gemm_kernel(const _Float16* __restrict__ X,   // [BS, D]
                const _Float16* __restrict__ Wm,  // [3D, D]
                const float*    __restrict__ bias,// [3D]
                _Float16*       __restrict__ QKV) // [BS, 3D]
{
  const int lane = threadIdx.x & 31;
  const int wave = threadIdx.x >> 5;
  const int row  = lane & 15;
  const int hi   = lane >> 4;
  const int tn   = blockIdx.x * 128 + (wave & 1) * 64;
  const int tm   = blockIdx.y * 128 + (wave >> 1) * 32;

  const _Float16* arow0 = X + (size_t)(tm + row) * DDIM;
  const _Float16* arow1 = arow0 + (size_t)16 * DDIM;
  const _Float16* brow[4];
#pragma unroll
  for (int t = 0; t < 4; ++t)
    brow[t] = Wm + (size_t)(tn + t * 16 + row) * DDIM;

  v8f acc[2][4] = {};
  for (int k0 = 0; k0 < DDIM; k0 += 32) {
    v16h a0 = load_frag_a_f16(arow0, k0, lane);
    v16h a1 = load_frag_a_f16(arow1, k0, lane);
    v16h bb[4];
#pragma unroll
    for (int t = 0; t < 4; ++t) bb[t] = load_frag_b_f16(brow[t], k0, lane);
#pragma unroll
    for (int t = 0; t < 4; ++t) {
      acc[0][t] = wmma_f16(a0, bb[t], acc[0][t]);
      acc[1][t] = wmma_f16(a1, bb[t], acc[1][t]);
    }
  }

#pragma unroll
  for (int mt = 0; mt < 2; ++mt) {
#pragma unroll
    for (int t = 0; t < 4; ++t) {
      const int n  = tn + t * 16 + row;
      const float bv = bias[n];
#pragma unroll
      for (int j = 0; j < 8; ++j) {
        const int m = tm + mt * 16 + j + hi * 8;
        QKV[(size_t)m * E3D + n] = (_Float16)(acc[mt][t][j] + bv);
      }
    }
  }
}

// ---------------------------------------------------------------------------
// Kernel 3: LDS-tiled transpose  VT[b][d][j] = V[b][j][d]  (f16)
//   64x64 tiles, row padded to 66 halves (33-dword stride -> conflict free).
// ---------------------------------------------------------------------------
__global__ void __launch_bounds__(256)
transpose_v_kernel(const _Float16* __restrict__ QKV, _Float16* __restrict__ VT) {
  __shared__ _Float16 tile[64][66];
  const int b  = blockIdx.z;
  const int d0 = blockIdx.x * 64;
  const int j0 = blockIdx.y * 64;
  const int c  = threadIdx.x & 63;
  const int r0 = threadIdx.x >> 6;              // 0..3

#pragma unroll
  for (int it = 0; it < 16; ++it) {
    const int jr = r0 + it * 4;                 // 0..63
    tile[jr][c] = QKV[((size_t)(b * SDIM + j0 + jr)) * E3D + 2048 + d0 + c];
  }
  __syncthreads();
#pragma unroll
  for (int it = 0; it < 16; ++it) {
    const int dr = r0 + it * 4;                 // 0..63
    VT[((size_t)(b * DDIM + d0 + dr)) * SDIM + j0 + c] = tile[c][dr];
  }
}

// ---------------------------------------------------------------------------
// Kernel 4: scores[b][i][j] = (1/32) * sum_d q[b,i,d] * k[b,j,d]  (fp32 out)
//   Per batch M = N = 2048, K = 1024.  Wave 32x64, block 128x128.
// ---------------------------------------------------------------------------
__global__ void __launch_bounds__(256)
scores_gemm_kernel(const _Float16* __restrict__ QKV, float* __restrict__ Wout) {
  const int lane = threadIdx.x & 31;
  const int wave = threadIdx.x >> 5;
  const int row  = lane & 15;
  const int hi   = lane >> 4;
  const int b    = blockIdx.z;
  const int tn   = blockIdx.x * 128 + (wave & 1) * 64;   // key tile
  const int tm   = blockIdx.y * 128 + (wave >> 1) * 32;  // query tile

  const _Float16* base  = QKV + (size_t)b * SDIM * E3D;
  const _Float16* arow0 = base + (size_t)(tm + row) * E3D;        // q rows
  const _Float16* arow1 = arow0 + (size_t)16 * E3D;
  const _Float16* brow[4];
#pragma unroll
  for (int t = 0; t < 4; ++t)
    brow[t] = base + (size_t)(tn + t * 16 + row) * E3D + DDIM;    // k rows

  v8f acc[2][4] = {};
  for (int k0 = 0; k0 < DDIM; k0 += 32) {
    v16h a0 = load_frag_a_f16(arow0, k0, lane);
    v16h a1 = load_frag_a_f16(arow1, k0, lane);
    v16h bb[4];
#pragma unroll
    for (int t = 0; t < 4; ++t) bb[t] = load_frag_b_f16(brow[t], k0, lane);
#pragma unroll
    for (int t = 0; t < 4; ++t) {
      acc[0][t] = wmma_f16(a0, bb[t], acc[0][t]);
      acc[1][t] = wmma_f16(a1, bb[t], acc[1][t]);
    }
  }

#pragma unroll
  for (int mt = 0; mt < 2; ++mt) {
#pragma unroll
    for (int t = 0; t < 4; ++t) {
      const int n = tn + t * 16 + row;
#pragma unroll
      for (int j = 0; j < 8; ++j) {
        const int m = tm + mt * 16 + j + hi * 8;
        Wout[((size_t)(b * SDIM + m)) * SDIM + n] = acc[mt][t][j] * INV_SCALE;
      }
    }
  }
}

// ---------------------------------------------------------------------------
// Kernel 5: row-wise softmax in place over scores (fp32), one block per row
// ---------------------------------------------------------------------------
__global__ void __launch_bounds__(256)
softmax_kernel(float* __restrict__ Wmat) {
  float* row = Wmat + (size_t)blockIdx.x * SDIM;
  __shared__ float red[256];
  const int t = threadIdx.x;

  float mx = -3.402823466e+38f;
  for (int j = t; j < SDIM; j += 256) mx = fmaxf(mx, row[j]);
  red[t] = mx; __syncthreads();
  for (int s = 128; s > 0; s >>= 1) {
    if (t < s) red[t] = fmaxf(red[t], red[t + s]);
    __syncthreads();
  }
  mx = red[0]; __syncthreads();

  float sum = 0.0f;
  for (int j = t; j < SDIM; j += 256) {
    float e = __expf(row[j] - mx);
    row[j] = e;
    sum += e;
  }
  red[t] = sum; __syncthreads();
  for (int s = 128; s > 0; s >>= 1) {
    if (t < s) red[t] += red[t + s];
    __syncthreads();
  }
  const float inv = 1.0f / red[0];

  for (int j = t; j < SDIM; j += 256) row[j] *= inv;
}

// ---------------------------------------------------------------------------
// Kernel 6: out[b][i][d] = sum_j w[b,i,j] * v[b,j,d]
//   A = fp32 softmax weights (converted in-register), B = VT (f16).
//   Per batch M = 2048, N = 1024, K = 2048.  Wave 32x64, block 128x128.
// ---------------------------------------------------------------------------
__global__ void __launch_bounds__(256)
av_gemm_kernel(const float* __restrict__ Wmat,    // [B,S,S] fp32
               const _Float16* __restrict__ VT,   // [B,D,S] f16
               float* __restrict__ Out)           // [B,S,D] fp32
{
  const int lane = threadIdx.x & 31;
  const int wave = threadIdx.x >> 5;
  const int row  = lane & 15;
  const int hi   = lane >> 4;
  const int b    = blockIdx.z;
  const int tn   = blockIdx.x * 128 + (wave & 1) * 64;   // d tile
  const int tm   = blockIdx.y * 128 + (wave >> 1) * 32;  // query tile

  const float* arow0 = Wmat + ((size_t)(b * SDIM + tm + row)) * SDIM;
  const float* arow1 = arow0 + (size_t)16 * SDIM;
  const _Float16* brow[4];
#pragma unroll
  for (int t = 0; t < 4; ++t)
    brow[t] = VT + ((size_t)(b * DDIM + tn + t * 16 + row)) * SDIM;

  v8f acc[2][4] = {};
  for (int k0 = 0; k0 < SDIM; k0 += 32) {
    v16h a0 = load_frag_a_f32(arow0, k0, lane);
    v16h a1 = load_frag_a_f32(arow1, k0, lane);
    v16h bb[4];
#pragma unroll
    for (int t = 0; t < 4; ++t) bb[t] = load_frag_b_f16(brow[t], k0, lane);
#pragma unroll
    for (int t = 0; t < 4; ++t) {
      acc[0][t] = wmma_f16(a0, bb[t], acc[0][t]);
      acc[1][t] = wmma_f16(a1, bb[t], acc[1][t]);
    }
  }

#pragma unroll
  for (int mt = 0; mt < 2; ++mt) {
#pragma unroll
    for (int t = 0; t < 4; ++t) {
      const int n = tn + t * 16 + row;
#pragma unroll
      for (int j = 0; j < 8; ++j) {
        const int m = tm + mt * 16 + j + hi * 8;
        Out[((size_t)(b * SDIM + m)) * DDIM + n] = acc[mt][t][j];
      }
    }
  }
}

// ---------------------------------------------------------------------------
extern "C" void kernel_launch(void* const* d_in, const int* in_sizes, int n_in,
                              void* d_out, int out_size, void* d_ws, size_t ws_size,
                              hipStream_t stream) {
  (void)in_sizes; (void)n_in; (void)out_size; (void)ws_size;

  const float* x    = (const float*)d_in[0];   // [B,S,D]
  const float* Wqkv = (const float*)d_in[1];   // [3D,D]
  const float* bqkv = (const float*)d_in[2];   // [3D]

  float* out  = (float*)d_out;                     // [B,S,D]
  float* wmat = out + (size_t)BDIM * SDIM * DDIM;  // [B,S,S]

  _Float16* x16   = (_Float16*)d_ws;                    // 16.8 MB
  _Float16* w16   = x16  + (size_t)BS * DDIM;           //  6.3 MB
  _Float16* qkv16 = w16  + (size_t)E3D * DDIM;          // 50.3 MB
  _Float16* vt16  = qkv16 + (size_t)BS * E3D;           // 16.8 MB

  const int nx = BS * DDIM;        // 8,388,608
  const int nw = E3D * DDIM;       // 3,145,728

  cvt_f32_to_f16_kernel<<<(nx + 255) / 256, 256, 0, stream>>>(x, x16, nx);
  cvt_f32_to_f16_kernel<<<(nw + 255) / 256, 256, 0, stream>>>(Wqkv, w16, nw);

  // QKV projection: M=8192, N=3072, K=1024 ; block tile 128x128
  qkv_gemm_kernel<<<dim3(E3D / 128, BS / 128), 256, 0, stream>>>(x16, w16, bqkv, qkv16);

  // LDS-tiled transpose of V -> [B, D, S]
  transpose_v_kernel<<<dim3(DDIM / 64, SDIM / 64, BDIM), 256, 0, stream>>>(qkv16, vt16);

  // Scores: per batch M=N=2048, K=1024 -> fp32 into d_out's w region
  scores_gemm_kernel<<<dim3(SDIM / 128, SDIM / 128, BDIM), 256, 0, stream>>>(qkv16, wmat);

  // Softmax in place (fp32)
  softmax_kernel<<<BS, 256, 0, stream>>>(wmat);

  // AV: per batch M=2048, N=1024, K=2048
  av_gemm_kernel<<<dim3(DDIM / 128, SDIM / 128, BDIM), 256, 0, stream>>>(wmat, vt16, out);
}
